// LinearWDecoder_34746285424672
// MI455X (gfx1250) — matrix operand order repaired and checked by reading, not compiled, not run on hardware
//
#include <hip/hip_runtime.h>

// ---------------------------------------------------------------------------
// Grouped MLP decoder, fused: z -> L1(relu) -> L2(relu)+res -> L3(relu)+res -> Lo
// bf16 WMMA (v_wmma_f32_16x16x32_bf16), f32 accumulate, activations resident in
// LDS, weights double-buffered in LDS via GLOBAL_LOAD_ASYNC_TO_LDS_B128.
// ---------------------------------------------------------------------------

typedef __bf16 v16bf __attribute__((ext_vector_type(16)));
typedef __bf16 v8bf  __attribute__((ext_vector_type(8)));
typedef float  v8f   __attribute__((ext_vector_type(8)));

static constexpr int Bq  = 8192;
static constexpr int Gq  = 16;
static constexpr int CIN = 128;
static constexpr int Hq  = 256;
static constexpr int PDq = 256;

static constexpr int BM = 128;   // rows per workgroup tile
static constexpr int BN = 256;   // full output width per group
static constexpr int KC = 64;    // K-chunk staged in LDS
static constexpr int AP = 264;   // act LDS pitch (bf16) : 528B = 33*16, conflict-free
static constexpr int WP = 72;    // weight LDS pitch (bf16): 144B = 9*16

static constexpr int WBUF_ELEMS = BN * WP;                 // one weight buffer
static constexpr size_t SMEM_BYTES =
    (size_t)(BM * AP + 2 * WBUF_ELEMS) * sizeof(unsigned short);   // 141312

__device__ __forceinline__ unsigned short f2bf(float f) {
  unsigned int u = __float_as_uint(f);
  unsigned int r = (u + 0x7FFFu + ((u >> 16) & 1u)) >> 16;   // RNE
  return (unsigned short)r;
}
__device__ __forceinline__ float bf2f(unsigned short h) {
  return __uint_as_float(((unsigned int)h) << 16);
}

// LDS byte offset of a generic pointer into shared memory (low 32 bits carry
// the DS-space address, identical to what ds_load_* already consumes).
__device__ __forceinline__ unsigned int lds_off(const void* p) {
  return (unsigned int)(size_t)p;
}

template <int N>
__device__ __forceinline__ void wait_asynccnt() {
#if __has_builtin(__builtin_amdgcn_s_wait_asynccnt)
  __builtin_amdgcn_s_wait_asynccnt(N);
#else
  asm volatile("s_wait_asynccnt %0" :: "n"(N) : "memory");
#endif
}

// Issue one K-chunk of weights (256 rows x 64 k, bf16) into LDS asynchronously.
// Wsrc points at W[0][k0]; per thread: 8 x b128 (16B) transfers, coalesced.
__device__ __forceinline__ void issue_wchunk(const unsigned short* Wsrc, int K,
                                             unsigned int buf_off, int tid) {
  #pragma unroll
  for (int r = 0; r < 8; ++r) {
    int u  = r * 256 + tid;
    int n  = u >> 3;             // output row 0..255
    int k8 = (u & 7) * 8;        // k offset within chunk, 8-elem units
    unsigned int loff = buf_off + (unsigned int)(n * WP + k8) * 2u;
    unsigned int voff = (unsigned int)(n * K + k8) * 2u;
    asm volatile("global_load_async_to_lds_b128 %0, %1, %2 offset:0"
                 :: "v"(loff), "v"(voff), "s"(Wsrc) : "memory");
  }
}

// --- prep: convert fp32 weights to bf16 in workspace (stays hot in L2) ---
__global__ void cvt_w_kernel(const float* __restrict__ src,
                             unsigned short* __restrict__ dst, int n) {
  int i = blockIdx.x * blockDim.x + threadIdx.x;
  int stride = gridDim.x * blockDim.x;
  for (; i < n; i += stride) dst[i] = f2bf(src[i]);
}

__global__ __launch_bounds__(256, 1)
void decoder_fused_kernel(const float* __restrict__ z1,
                          const float* __restrict__ z2,
                          const unsigned short* __restrict__ wq,   // bf16: W1|W2|W3|Wo
                          const float* __restrict__ b1,
                          const float* __restrict__ b2,
                          const float* __restrict__ b3,
                          const float* __restrict__ bo,
                          float* __restrict__ out) {
  extern __shared__ unsigned short smem[];
  unsigned short* act = smem;                 // [BM][AP] bf16 activation tile
  unsigned short* wls = smem + BM * AP;       // 2 x [BN][WP] bf16 weight buffers
  const unsigned int wls_base = lds_off(wls);

  const int g    = blockIdx.x & (Gq - 1);
  const int m0   = (blockIdx.x >> 4) * BM;
  const int tid  = threadIdx.x;
  const int lane = tid & 31;
  const int wave = tid >> 5;
  const int wm   = wave >> 2;      // 0..1 : 64-row band
  const int wn   = wave & 3;       // 0..3 : 64-col band
  const int hl   = lane >> 4;      // half-wave select
  const int lr   = lane & 15;

  // ---- stage z tile: cols 0..63 from z1[:,g], 64..127 from z2[:,g] ----
  {
    const float* z1g = z1 + (size_t)g * 64;
    const float* z2g = z2 + (size_t)g * 64;
    for (int u = tid; u < BM * 32; u += 256) {      // 32 float4 per row
      int m  = u >> 5;
      int c4 = (u & 31) * 4;
      const float* src = (c4 < 64)
          ? (z1g + (size_t)(m0 + m) * (Gq * 64) + c4)
          : (z2g + (size_t)(m0 + m) * (Gq * 64) + (c4 - 64));
      float4 v = *(const float4*)src;
      unsigned short* d = act + m * AP + c4;
      d[0] = f2bf(v.x); d[1] = f2bf(v.y); d[2] = f2bf(v.z); d[3] = f2bf(v.w);
    }
  }

  const size_t w1off = 0;
  const size_t w2off = (size_t)Gq * Hq * CIN;
  const size_t w3off = w2off + (size_t)Gq * Hq * Hq;
  const size_t wooff = w3off + (size_t)Gq * Hq * Hq;

  for (int layer = 0; layer < 4; ++layer) {
    int K; const unsigned short* W; const float* bias;
    switch (layer) {
      case 0:  K = CIN; W = wq + w1off + (size_t)g * Hq * CIN; bias = b1 + g * Hq;  break;
      case 1:  K = Hq;  W = wq + w2off + (size_t)g * Hq * Hq;  bias = b2 + g * Hq;  break;
      case 2:  K = Hq;  W = wq + w3off + (size_t)g * Hq * Hq;  bias = b3 + g * Hq;  break;
      default: K = Hq;  W = wq + wooff + (size_t)g * PDq * Hq; bias = bo + g * PDq; break;
    }
    const int nc = K / KC;                 // 2 (layer 0) or 4 chunks

    v8f acc[4][4] = {};   // 4x4 fragments of 16x16 f32 = 64x64 per wave

    // prologue: async-stage chunk 0 into buffer 0
    issue_wchunk(W, K, wls_base, tid);

    for (int c = 0; c < nc; ++c) {
      const int cur = c & 1;
      if (c + 1 < nc)   // prefetch next chunk into the other buffer
        issue_wchunk(W + (c + 1) * KC, K, wls_base + (cur ^ 1) * (WBUF_ELEMS * 2), tid);
      // in-order completion: oldest 8 (current buffer) done when cnt <= 8
      if (c + 1 < nc) wait_asynccnt<8>(); else wait_asynccnt<0>();
      __syncthreads();                      // current buffer visible to all waves

      const unsigned short* wcur = wls + cur * WBUF_ELEMS;
      const int k0 = c * KC;
      #pragma unroll
      for (int ks = 0; ks < KC; ks += 32) {
        // A fragments: 16x32 bf16; lane<16: K {ks..+7, ks+16..+23}; lane>=16: +8
        v16bf afrag[4];
        #pragma unroll
        for (int i = 0; i < 4; ++i) {
          int m = wm * 64 + i * 16 + lr;
          const unsigned short* base = act + m * AP + k0 + ks + hl * 8;
          union { v16bf v; struct { v8bf lo, hi; } s; } ua;
          ua.s.lo = *(const v8bf*)(base);
          ua.s.hi = *(const v8bf*)(base + 16);   // +16 K positions
          afrag[i] = ua.v;
        }
        // B fragments: 32x16 bf16; lane<16: n=lr, K {ks..+7, ks+16..+23}; lane>=16: +8
        v16bf bfrag[4];
        #pragma unroll
        for (int j = 0; j < 4; ++j) {
          int n = wn * 64 + j * 16 + lr;
          const unsigned short* base = wcur + n * WP + ks + hl * 8;
          union { v16bf v; struct { v8bf lo, hi; } s; } ub;
          ub.s.lo = *(const v8bf*)(base);
          ub.s.hi = *(const v8bf*)(base + 16);
          bfrag[j] = ub.v;
        }
        #pragma unroll
        for (int i = 0; i < 4; ++i)
          #pragma unroll
          for (int j = 0; j < 4; ++j)
            acc[i][j] = __builtin_amdgcn_wmma_f32_16x16x32_bf16(
                false, afrag[i], false, bfrag[j],
                (short)0, acc[i][j], false, false);
      }
      __syncthreads();   // all waves done reading this buffer (and, on the last
                         // chunk, done reading act) before it is re-targeted
    }

    const bool relu  = (layer < 3);
    const bool resid = (layer == 1 || layer == 2);
    const bool last  = (layer == 3);

    #pragma unroll
    for (int j = 0; j < 4; ++j) {
      int n = wn * 64 + j * 16 + lr;
      float bv = bias[n];
      #pragma unroll
      for (int i = 0; i < 4; ++i) {
        int mb = wm * 64 + i * 16 + hl * 8;
        #pragma unroll
        for (int v = 0; v < 8; ++v) {
          int m = mb + v;                      // C/D layout: vgpr=v, lane -> (m,n)
          float x = acc[i][j][v] + bv;
          if (relu)  x = fmaxf(x, 0.0f);
          if (resid) x += bf2f(act[m * AP + n]);   // own element: no cross-thread hazard
          if (last)
            out[(size_t)(m0 + m) * (Gq * PDq) + (size_t)g * PDq + n] = x;
          else
            act[m * AP + n] = f2bf(x);
        }
      }
    }
    // next layer's first post-issue barrier orders these writes vs. future reads
  }
}

extern "C" void kernel_launch(void* const* d_in, const int* in_sizes, int n_in,
                              void* d_out, int out_size, void* d_ws, size_t ws_size,
                              hipStream_t stream) {
  (void)in_sizes; (void)n_in; (void)out_size; (void)ws_size;
  const float* z1 = (const float*)d_in[0];
  const float* z2 = (const float*)d_in[1];
  const float* W1 = (const float*)d_in[2];
  const float* b1 = (const float*)d_in[3];
  const float* W2 = (const float*)d_in[4];
  const float* b2 = (const float*)d_in[5];
  const float* W3 = (const float*)d_in[6];
  const float* b3 = (const float*)d_in[7];
  const float* Wo = (const float*)d_in[8];
  const float* bo = (const float*)d_in[9];
  float* out = (float*)d_out;

  unsigned short* wq = (unsigned short*)d_ws;
  const int n1 = Gq * Hq * CIN;   // 524288
  const int n2 = Gq * Hq * Hq;    // 1048576
  cvt_w_kernel<<<1024, 256, 0, stream>>>(W1, wq, n1);
  cvt_w_kernel<<<1024, 256, 0, stream>>>(W2, wq + (size_t)n1, n2);
  cvt_w_kernel<<<1024, 256, 0, stream>>>(W3, wq + (size_t)n1 + n2, n2);
  cvt_w_kernel<<<1024, 256, 0, stream>>>(Wo, wq + (size_t)n1 + 2 * (size_t)n2, n2);

  (void)hipFuncSetAttribute((const void*)decoder_fused_kernel,
                            hipFuncAttributeMaxDynamicSharedMemorySize,
                            (int)SMEM_BYTES);
  const int grid = (Bq / BM) * Gq;   // 64 row-tiles * 16 groups = 1024
  decoder_fused_kernel<<<grid, 256, SMEM_BYTES, stream>>>(
      z1, z2, wq, b1, b2, b3, bo, out);
}
